// DynamicWideConvBlock_15685220565271
// MI455X (gfx1250) — matrix operand order repaired and checked by reading, not compiled, not run on hardware
//
#include <hip/hip_runtime.h>
#include <hip/hip_bf16.h>
#include <cstdint>

typedef __attribute__((ext_vector_type(8)))  float  v8f;
typedef __attribute__((ext_vector_type(16))) __bf16 v16bf;

namespace {
constexpr int NB = 4, NC = 64, NH = 256, NW = 256, NWIDE = 256, NK = 4, NRED = 16;
constexpr int HWp = NH * NW;   // 65536
constexpr int WP  = NW + 2;    // W padded with 1-pixel halo on both sides (zero)
}

// ---------------- weight-norm row norms: one block per row ----------------
__global__ void rownorm_kernel(const float* __restrict__ v, float* __restrict__ norm, int rowlen) {
    __shared__ float sd[256];
    const int row = blockIdx.x;
    float s = 0.f;
    for (int i = threadIdx.x; i < rowlen; i += 256) {
        float t = v[(long)row * rowlen + i];
        s += t * t;
    }
    sd[threadIdx.x] = s;
    __syncthreads();
    for (int o = 128; o > 0; o >>= 1) {
        if (threadIdx.x < o) sd[threadIdx.x] += sd[threadIdx.x + o];
        __syncthreads();
    }
    if (threadIdx.x == 0) norm[row] = sqrtf(sd[0]);
}

// ---------------- routing softmax + dynamic bias mix ----------------
__global__ void routing_kernel(const float* __restrict__ scale, const float* __restrict__ att_w,
                               const float* __restrict__ att_b, const float* __restrict__ dyn_b,
                               float* __restrict__ att, float* __restrict__ bias_b) {
    __shared__ float a[NB * NK];
    const int tid = threadIdx.x;
    if (tid < NB) {
        float sc = scale[tid];
        float l[NK]; float mx = -1e30f;
        for (int k = 0; k < NK; ++k) { l[k] = sc * att_w[k] + att_b[k]; mx = fmaxf(mx, l[k]); }
        float s = 0.f;
        for (int k = 0; k < NK; ++k) { l[k] = expf(l[k] - mx); s += l[k]; }
        for (int k = 0; k < NK; ++k) { float p = l[k] / s; a[tid * NK + k] = p; att[tid * NK + k] = p; }
    }
    __syncthreads();
    for (int i = tid; i < NB * NC; i += blockDim.x) {
        int b = i / NC, c = i % NC;
        float s = 0.f;
        for (int k = 0; k < NK; ++k) s += a[b * NK + k] * dyn_b[k * NC + c];
        bias_b[i] = s;
    }
}

// A fragment (16x32 bf16, wave32): lane<16: K in {0..15}; lane>=16: K in {16..31}
__device__ __forceinline__ int a_frag_k(int lane, int i) {
    return (lane < 16 ? 0 : 16) + i;
}

// ---------------- pack weight-normed static conv weights into WMMA A fragments ----------------
__global__ void pack_wn_kernel(const float* __restrict__ v, const float* __restrict__ g,
                               const float* __restrict__ norm, __bf16* __restrict__ out,
                               int COUT, int CIN) {
    const int total = COUT * CIN * 9;
    const int idx = blockIdx.x * 256 + threadIdx.x;
    if (idx >= total) return;
    const int nkc = CIN / 32;
    int j = idx & 511, f = idx >> 9;
    int lane = j >> 4, i = j & 15;
    int kc = f % nkc; f /= nkc;
    int t = f % 9; int mt = f / 9;
    int m = mt * 16 + (lane & 15);
    int cin = kc * 32 + a_frag_k(lane, i);
    float val = g[m] * v[((long)(m * CIN + cin)) * 9 + t] / norm[m];
    out[idx] = (__bf16)val;
}

// ---------------- pack per-sample dynamic weights (attention-mixed bank) ----------------
__global__ void pack_dyn_kernel(const float* __restrict__ dyn_v, const float* __restrict__ dyn_g,
                                const float* __restrict__ norm, const float* __restrict__ att,
                                __bf16* __restrict__ out) {
    const int per = NC * NC * 9;           // 36864
    const int idx = blockIdx.x * 256 + threadIdx.x;
    if (idx >= NB * per) return;
    const int b = idx / per;
    const int e = idx % per;
    const int nkc = NC / 32;
    int j = e & 511, f = e >> 9;
    int lane = j >> 4, i = j & 15;
    int kc = f % nkc; f /= nkc;
    int t = f % 9; int mt = f / 9;
    int m = mt * 16 + (lane & 15);
    int cin = kc * 32 + a_frag_k(lane, i);
    long wi = ((long)(m * NC + cin)) * 9 + t;
    float s = 0.f;
    for (int k = 0; k < NK; ++k)
        s += att[b * NK + k] * dyn_g[k] * dyn_v[(long)k * per + wi] / norm[k];
    out[idx] = (__bf16)s;
}

// ---------------- x (fp32 NCHW) -> bf16 NHWC (halo-padded in W) ----------------
__global__ void to_bf16_nhwc_kernel(const float* __restrict__ x, __bf16* __restrict__ xh) {
    const long idx = (long)blockIdx.x * 256 + threadIdx.x;   // (b*HW + hw)*64 + c
    const int  c   = (int)(idx & 63);
    const long p   = idx >> 6;
    const int  b   = (int)(p >> 16);
    const int  hw  = (int)(p & 65535);
    const int  hh  = hw >> 8, ww = hw & 255;
    xh[(((long)b * NH + hh) * WP + 1 + ww) * NC + c] = (__bf16)x[((long)b * NC + c) * HWp + hw];
}

// zero the two halo columns of a [NB][NH][WP][Cn] bf16 buffer
__global__ void zero_halo_kernel(__bf16* __restrict__ buf, int Cn) {
    const int idx = blockIdx.x * 256 + threadIdx.x;
    const int total = NB * NH * 2 * Cn;
    if (idx >= total) return;
    const int c = idx % Cn;
    const int r = idx / Cn;
    const int side = r & 1;
    const int row = r >> 1;                       // [0, NB*NH)
    buf[((long)row * WP + (side ? (WP - 1) : 0)) * Cn + c] = (__bf16)0.0f;
}

__global__ void zero_f32_kernel(float* p, int n) {
    int i = blockIdx.x * 256 + threadIdx.x;
    if (i < n) p[i] = 0.f;
}

// ---------------- implicit-GEMM 3x3 conv via bf16 WMMA ----------------
// in:    [B][NH][WP][CIN] bf16, halo columns zero -> all B loads are unconditional.
// wfrag: [COUT/16][9][CIN/32][512] bf16 A-fragments (+ per-batch stride for dyn conv)
// Block = 8 waves; each wave: 16 cout x 64 pixels (4 interleaved accumulator chains).
// Weights staged into LDS by async DMA (global_load_async_to_lds_b128 + s_wait_asynccnt):
// all 9 taps at once when <=32KB, else one ky-row (3 taps) at a time.
// OUT_MODE 0: bf16 NHWC(padded) out (+bias, optional ReLU)
// OUT_MODE 1: f32 NCHW out (+per-sample bias) and channel-sum pooling via shuffle + atomics
template<int CIN, int COUT, bool RELU, int OUT_MODE>
__global__ __launch_bounds__(256)
void conv3x3_wmma_kernel(const __bf16* __restrict__ in, const __bf16* __restrict__ wfrag,
                         const float* __restrict__ bias,
                         __bf16* __restrict__ outb, float* __restrict__ outf,
                         float* __restrict__ pool, long wstride, int bstride) {
    constexpr int NKC = CIN / 32;                 // K-chunks per tap
    constexpr int NFRAG = 9 * NKC;                // fragments per cout-tile
    constexpr bool ONESHOT = (NFRAG * 1024) <= 32768;
    constexpr int SFRAGS = ONESHOT ? NFRAG : 3 * NKC;
    __shared__ alignas(128) __bf16 sA[SFRAGS * 512];   // 18KB (oneshot) or 24KB (conv2)

    const int tid  = threadIdx.x;
    const int lane = tid & 31;
    const int wave = tid >> 5;
    const int z    = blockIdx.z;
    const int mt   = blockIdx.y;

    // block covers 2 rows x 256 cols: waves 0-3 -> row h0 (64-col segments), waves 4-7 -> row h0+1
    const int h   = blockIdx.x * 2 + (wave >> 2);
    const int w0  = (wave & 3) * 64;
    const int n   = lane & 15;                    // N index (pixel within 16-group)
    const int hiK = (lane < 16) ? 0 : 16;         // B-fragment K offset per half-wave
    const __bf16* inB = in + (long)z * NH * WP * CIN;
    const __bf16* wF  = wfrag + (long)z * wstride + (long)mt * (NFRAG * 512);
    const uint32_t ldsBase = (uint32_t)(uintptr_t)(void*)sA;

    auto stage = [&](int fragBase, int nfrags) {
        const char* gsrc = (const char*)(wF + (long)fragBase * 512);
        const int chunks = nfrags * 64;           // 16B chunks
        for (int c4 = tid; c4 < chunks; c4 += 256) {
            const uint32_t loff = ldsBase + c4 * 16;
            const void* gp = gsrc + (long)c4 * 16;
            asm volatile("global_load_async_to_lds_b128 %0, %1, off"
                         :: "v"(loff), "v"(gp) : "memory");
        }
        asm volatile("s_wait_asynccnt 0" ::: "memory");
        __syncthreads();
    };

    if constexpr (ONESHOT) stage(0, NFRAG);

    const v8f zero8 = {0.f, 0.f, 0.f, 0.f, 0.f, 0.f, 0.f, 0.f};
    v8f a0 = zero8, a1 = zero8, a2 = zero8, a3 = zero8;

    for (int ky = 0; ky < 3; ++ky) {
        if constexpr (!ONESHOT) stage(ky * 3 * NKC, 3 * NKC);
        const int hy = h + ky - 1;
        if (hy >= 0 && hy < NH) {                 // uniform per wave; no barrier inside
            // one per-lane base address for this row; everything else is immediate offsets
            const __bf16* pb = inB + ((long)hy * WP + (w0 + n + 1)) * CIN + hiK;
            const int fb = ONESHOT ? ky * 3 * NKC : 0;
#pragma unroll
            for (int kx = 0; kx < 3; ++kx) {
#pragma unroll
                for (int kc = 0; kc < NKC; ++kc) {
                    const v16bf afr = *(const v16bf*)(sA + (fb + kx * NKC + kc) * 512 + lane * 16);
                    const long o = (long)(kx - 1) * CIN + kc * 32;
                    const v16bf b0 = *(const v16bf*)(pb + o);
                    const v16bf b1 = *(const v16bf*)(pb + o + 16 * CIN);
                    const v16bf b2 = *(const v16bf*)(pb + o + 32 * CIN);
                    const v16bf b3 = *(const v16bf*)(pb + o + 48 * CIN);
                    a0 = __builtin_amdgcn_wmma_f32_16x16x32_bf16(false, afr, false, b0, (short)0, a0, false, false);
                    a1 = __builtin_amdgcn_wmma_f32_16x16x32_bf16(false, afr, false, b1, (short)0, a1, false, false);
                    a2 = __builtin_amdgcn_wmma_f32_16x16x32_bf16(false, afr, false, b2, (short)0, a2, false, false);
                    a3 = __builtin_amdgcn_wmma_f32_16x16x32_bf16(false, afr, false, b3, (short)0, a3, false, false);
                }
            }
        }
        if constexpr (!ONESHOT) __syncthreads();  // sA reused next ky-row
    }

    const int mBase = mt * 16 + (lane >> 4) * 8;  // D layout: VGPR r -> M = (lane/16)*8 + r
    const float* bptr = bias + (long)z * bstride + mBase;
    float bv[8];
#pragma unroll
    for (int r = 0; r < 8; ++r) bv[r] = bptr[r];

    v8f acc[4] = {a0, a1, a2, a3};
    if (OUT_MODE == 0) {
#pragma unroll
        for (int nt = 0; nt < 4; ++nt) {
            __bf16* op = outb + (((long)z * NH + h) * WP + 1 + w0 + nt * 16 + n) * COUT + mBase;
            union { __bf16 hv[8]; int4 q; } u;
#pragma unroll
            for (int r = 0; r < 8; ++r) {
                float v = acc[nt][r] + bv[r];
                if (RELU) v = fmaxf(v, 0.f);
                u.hv[r] = (__bf16)v;
            }
            *(int4*)op = u.q;                     // 16B coalesced store of 8 channels
        }
    } else {
#pragma unroll
        for (int nt = 0; nt < 4; ++nt) {
#pragma unroll
            for (int r = 0; r < 8; ++r) {
                const int c = mBase + r;
                float v = acc[nt][r] + bv[r];
                outf[(((long)z * COUT + c) * NH + h) * NW + w0 + nt * 16 + n] = v;
                float s = v;                      // reduce over 16 pixels (half-wave)
                s += __shfl_xor(s, 1, 32);
                s += __shfl_xor(s, 2, 32);
                s += __shfl_xor(s, 4, 32);
                s += __shfl_xor(s, 8, 32);
                if ((lane & 15) == 0) atomicAdd(&pool[z * COUT + c], s);
            }
        }
    }
}

// ---------------- SE gate from pooled sums ----------------
__global__ void se_gate_kernel(const float* __restrict__ pool, const float* __restrict__ w1,
                               const float* __restrict__ w2, float* __restrict__ gate) {
    __shared__ float mean[NC];
    __shared__ float hid[NRED];
    const int b = blockIdx.x, tid = threadIdx.x;  // blockDim == NC
    mean[tid] = pool[b * NC + tid] * (1.0f / (float)(NH * NW));
    __syncthreads();
    if (tid < NRED) {
        float s = 0.f;
        for (int c = 0; c < NC; ++c) s += mean[c] * w1[tid * NC + c];
        hid[tid] = fmaxf(s, 0.f);
    }
    __syncthreads();
    float s = 0.f;
    for (int r = 0; r < NRED; ++r) s += hid[r] * w2[tid * NRED + r];
    gate[b * NC + tid] = 1.f / (1.f + expf(-s));
}

// ---------------- out = x + y * gate ----------------
__global__ void finalize_kernel(const float* __restrict__ x, const float* __restrict__ y,
                                const float* __restrict__ gate, float* __restrict__ out) {
    const long i = (long)blockIdx.x * 256 + threadIdx.x;
    const int c = (int)((i >> 16) & 63);   // HW = 2^16, NC = 64
    const int b = (int)(i >> 22);          // NC*HW = 2^22
    out[i] = x[i] + y[i] * gate[b * NC + c];
}

extern "C" void kernel_launch(void* const* d_in, const int* in_sizes, int n_in,
                              void* d_out, int out_size, void* d_ws, size_t ws_size,
                              hipStream_t stream) {
    const float* x     = (const float*)d_in[0];
    const float* scale = (const float*)d_in[1];
    const float* v1    = (const float*)d_in[2];
    const float* g1    = (const float*)d_in[3];
    const float* b1    = (const float*)d_in[4];
    const float* v2    = (const float*)d_in[5];
    const float* g2    = (const float*)d_in[6];
    const float* b2    = (const float*)d_in[7];
    const float* dyn_v = (const float*)d_in[8];
    const float* dyn_g = (const float*)d_in[9];
    const float* dyn_b = (const float*)d_in[10];
    const float* att_w = (const float*)d_in[11];
    const float* att_b = (const float*)d_in[12];
    const float* se_w1 = (const float*)d_in[13];
    const float* se_w2 = (const float*)d_in[14];
    float* out = (float*)d_out;

    char* ws = (char*)d_ws;
    size_t off = 0;
    auto alloc = [&](size_t bytes) -> void* {
        void* p = ws + off;
        off = (off + bytes + 255) & ~(size_t)255;
        return p;
    };
    __bf16* xh   = (__bf16*)alloc((size_t)NB * NH * WP * NC * 2);      // 33.8 MB (padded)
    __bf16* r1   = (__bf16*)alloc((size_t)NB * NH * WP * NWIDE * 2);   // 135 MB (padded)
    __bf16* r2   = (__bf16*)alloc((size_t)NB * NH * WP * NC * 2);      // 33.8 MB (padded)
    float*  y    = (float*) alloc((size_t)NB * NC * HWp * 4);          // 67 MB
    __bf16* w1p  = (__bf16*)alloc((size_t)NWIDE * NC * 9 * 2);
    __bf16* w2p  = (__bf16*)alloc((size_t)NC * NWIDE * 9 * 2);
    __bf16* wdp  = (__bf16*)alloc((size_t)NB * NC * NC * 9 * 2);
    float* norm1 = (float*)alloc(NWIDE * 4);
    float* norm2 = (float*)alloc(NC * 4);
    float* normd = (float*)alloc(NK * 4);
    float* att   = (float*)alloc(NB * NK * 4);
    float* biasb = (float*)alloc(NB * NC * 4);
    float* pool  = (float*)alloc(NB * NC * 4);
    float* gate  = (float*)alloc(NB * NC * 4);

    // --- prep ---
    rownorm_kernel<<<NWIDE, 256, 0, stream>>>(v1, norm1, NC * 9);
    rownorm_kernel<<<NC,    256, 0, stream>>>(v2, norm2, NWIDE * 9);
    rownorm_kernel<<<NK,    256, 0, stream>>>(dyn_v, normd, NC * NC * 9);
    routing_kernel<<<1, 256, 0, stream>>>(scale, att_w, att_b, dyn_b, att, biasb);
    pack_wn_kernel<<<(NWIDE * NC * 9 + 255) / 256, 256, 0, stream>>>(v1, g1, norm1, w1p, NWIDE, NC);
    pack_wn_kernel<<<(NC * NWIDE * 9 + 255) / 256, 256, 0, stream>>>(v2, g2, norm2, w2p, NC, NWIDE);
    pack_dyn_kernel<<<(NB * NC * NC * 9 + 255) / 256, 256, 0, stream>>>(dyn_v, dyn_g, normd, att, wdp);
    to_bf16_nhwc_kernel<<<(NB * HWp * NC) / 256, 256, 0, stream>>>(x, xh);
    zero_halo_kernel<<<(NB * NH * 2 * NC + 255) / 256, 256, 0, stream>>>(xh, NC);
    zero_halo_kernel<<<(NB * NH * 2 * NWIDE + 255) / 256, 256, 0, stream>>>(r1, NWIDE);
    zero_halo_kernel<<<(NB * NH * 2 * NC + 255) / 256, 256, 0, stream>>>(r2, NC);
    zero_f32_kernel<<<1, 256, 0, stream>>>(pool, NB * NC);

    // --- WMMA implicit-GEMM conv stack (each block: 2 rows x 256 cols x 16 cout) ---
    dim3 grid1(NH / 2, NWIDE / 16, NB);   // 128 x 16 x 4
    conv3x3_wmma_kernel<NC, NWIDE, true, 0><<<grid1, 256, 0, stream>>>(
        xh, w1p, b1, r1, nullptr, nullptr, 0L, 0);
    dim3 grid2(NH / 2, NC / 16, NB);      // 128 x 4 x 4
    conv3x3_wmma_kernel<NWIDE, NC, false, 0><<<grid2, 256, 0, stream>>>(
        r1, w2p, b2, r2, nullptr, nullptr, 0L, 0);
    conv3x3_wmma_kernel<NC, NC, false, 1><<<grid2, 256, 0, stream>>>(
        r2, wdp, biasb, nullptr, y, pool, (long)(NC * NC * 9), NC);

    // --- SE gate + residual ---
    se_gate_kernel<<<NB, NC, 0, stream>>>(pool, se_w1, se_w2, gate);
    finalize_kernel<<<(NB * NC * HWp) / 256, 256, 0, stream>>>(x, y, gate, out);
}
